// SwarmByteRingModel_10857677324459
// MI455X (gfx1250) — matrix operand order repaired and checked by reading, not compiled, not run on hardware
//
#include <hip/hip_runtime.h>

typedef _Float16 v16h __attribute__((ext_vector_type(16)));
typedef _Float16 v8h  __attribute__((ext_vector_type(8)));
typedef _Float16 v4h  __attribute__((ext_vector_type(4)));
typedef float    v8f  __attribute__((ext_vector_type(8)));

#define NUM_BEINGS 32
#define NUM_BITS   256
#define MAX_H      1024
#define KMAX       128
#define BATCH      4096

// Being cycle (every 8): K = [128,64,32,16,8,4,2,2]
__constant__ int c_KS[NUM_BEINGS] = {
  128,64,32,16,8,4,2,2, 128,64,32,16,8,4,2,2,
  128,64,32,16,8,4,2,2, 128,64,32,16,8,4,2,2};
// H = clamp(8K, 128, 1024) rounded to 16
__constant__ int c_HS[NUM_BEINGS] = {
  1024,512,256,128,128,128,128,128, 1024,512,256,128,128,128,128,128,
  1024,512,256,128,128,128,128,128, 1024,512,256,128,128,128,128,128};
// Fibonacci tick periods by K rank
__constant__ int c_PER[NUM_BEINGS] = {
  21,13,8,5,3,2,1,1, 21,13,8,5,3,2,1,1,
  21,13,8,5,3,2,1,1, 21,13,8,5,3,2,1,1};

union Frag16 { v16h v; v8h h[2]; };

__device__ __forceinline__ float gelu_tanh(float v) {
  float u = v + 0.044715f * v * v * v;
  return 0.5f * v * (1.0f + tanhf(0.7978845608028654f * u));
}

// KPAD: compile-time padded contraction width (>=32 so the 16x16x32 WMMA
// K-depth is always covered; padded xg columns are zero -> exact result).
// KPAD=128 -> beings j*8; KPAD=64 -> j*8+1; KPAD=32 -> remaining 6 of each 8.
template <int KPAD>
__global__ __launch_bounds__(256)
void swarm_gemm(const float* __restrict__ x, const float* __restrict__ W1,
                const float* __restrict__ b1, const float* __restrict__ W2,
                const float* __restrict__ b2, const int* __restrict__ idx,
                const int* __restrict__ step_p, float* __restrict__ scat) {
  constexpr int KS32 = KPAD / 32;     // # 32-deep WMMA K steps
  constexpr int KT   = KPAD / 16;     // # 16-wide N tiles
  constexpr int XS   = KPAD + 8;      // LDS stride (halfs), 16B multiple
  constexpr int KSH  = (KPAD == 128) ? 7 : (KPAD == 64 ? 6 : 5);
  constexpr int WV   = KPAD / 32;     // float4 quads per thread for W staging
  constexpr int GQ   = KPAD / 8;      // gather quads per thread

  int being;
  if (KPAD == 128)      being = blockIdx.y * 8;
  else if (KPAD == 64)  being = blockIdx.y * 8 + 1;
  else                  being = (blockIdx.y / 6) * 8 + 2 + (blockIdx.y % 6);

  const int step = step_p[0];
  if (step % c_PER[being] != 0) return;          // uniform per block

  const int K = c_KS[being];
  const int H = c_HS[being];
  const int tid  = threadIdx.x;
  const int lane = tid & 31;
  const int wave = tid >> 5;
  const int m    = lane & 15;                    // row/col within tile
  const int hi   = lane >> 4;                    // half-wave select
  const int row0 = blockIdx.x * 128;

  extern __shared__ __align__(16) char smem[];
  int*      sidx = (int*)smem;                           // 512 B
  _Float16* xg   = (_Float16*)(smem + 512);              // 128*XS halfs
  _Float16* w1t  = xg  + 128 * XS;                       // 32*XS  ([h_local][k])
  _Float16* w2t  = w1t + 32 * XS;                        // KPAD*40 ([k][h_local])
  _Float16* hb   = w2t + KPAD * 40 + wave * (16 * 40);   // per-wave 16x40

  // ---- load gather indices ----
  for (int c = tid; c < KPAD; c += 256)
    sidx[c] = (c < K) ? idx[being * KMAX + c] : 0;
  __syncthreads();

  // ---- gather x columns into LDS (f16, zero-padded), 4 cols per step ----
  #pragma unroll
  for (int i = 0; i < GQ; ++i) {
    int q  = tid + i * 256;                 // quad index
    int r  = q >> (KSH - 2);
    int c4 = (q & (KPAD / 4 - 1)) * 4;
    v4h g;
    #pragma unroll
    for (int j = 0; j < 4; ++j) {
      int c = c4 + j;
      g[j] = (c < K) ? (_Float16)x[(row0 + r) * NUM_BITS + sidx[c]]
                     : (_Float16)0.0f;
    }
    *(v4h*)&xg[r * XS + c4] = g;            // ds_store_b64
  }
  __syncthreads();

  // ---- wave-resident 16-row A fragments, reused for all of H ----
  Frag16 A1[KS32];
  #pragma unroll
  for (int ks = 0; ks < KS32; ++ks) {
    const _Float16* p = xg + (wave * 16 + m) * XS + hi * 8 + ks * 32;
    A1[ks].h[0] = *(const v8h*)p;
    A1[ks].h[1] = *(const v8h*)(p + 16);
  }

  v8f acc[KT] = {};   // y accumulators: KT N-tiles of 16x16 f32 (registers)

  for (int hh = 0; hh < H; hh += 32) {
    // stage W1 chunk transposed: w1t[h_local][k], batched float4 reads
    {
      float4 t[WV];
      #pragma unroll
      for (int i = 0; i < WV; ++i) {
        int e = tid + i * 256;              // [0, KPAD*8)
        int k = e >> 3, h4 = (e & 7) * 4;
        t[i] = *(const float4*)&W1[(being * KMAX + k) * MAX_H + hh + h4];
      }
      #pragma unroll
      for (int i = 0; i < WV; ++i) {
        int e = tid + i * 256;
        int k = e >> 3, h4 = (e & 7) * 4;
        w1t[(h4 + 0) * XS + k] = (_Float16)t[i].x;
        w1t[(h4 + 1) * XS + k] = (_Float16)t[i].y;
        w1t[(h4 + 2) * XS + k] = (_Float16)t[i].z;
        w1t[(h4 + 3) * XS + k] = (_Float16)t[i].w;
      }
    }
    // stage W2 chunk transposed: w2t[k][h_local], batched float4 reads
    {
      float4 t[WV];
      #pragma unroll
      for (int i = 0; i < WV; ++i) {
        int e = tid + i * 256;              // [0, KPAD*8)
        int h = e >> (KSH - 2), k4 = (e & (KPAD / 4 - 1)) * 4;
        t[i] = *(const float4*)&W2[(being * MAX_H + hh + h) * KMAX + k4];
      }
      #pragma unroll
      for (int i = 0; i < WV; ++i) {
        int e = tid + i * 256;
        int h = e >> (KSH - 2), k4 = (e & (KPAD / 4 - 1)) * 4;
        w2t[(k4 + 0) * 40 + h] = (_Float16)t[i].x;
        w2t[(k4 + 1) * 40 + h] = (_Float16)t[i].y;
        w2t[(k4 + 2) * 40 + h] = (_Float16)t[i].z;
        w2t[(k4 + 3) * 40 + h] = (_Float16)t[i].w;
      }
    }
    if (hh + 32 < H)   // hint next W1 chunk into cache
      __builtin_prefetch(&W1[(being * KMAX) * MAX_H + hh + 32], 0, 1);
    __syncthreads();

    // ---- GEMM1 + gelu: two 16-col chunks -> 16x32 h tile in per-wave LDS ----
    #pragma unroll
    for (int chunk = 0; chunk < 2; ++chunk) {
      v8f c1 = {};
      #pragma unroll
      for (int ks = 0; ks < KS32; ++ks) {
        Frag16 B1;
        const _Float16* p = w1t + (chunk * 16 + m) * XS + hi * 8 + ks * 32;
        B1.h[0] = *(const v8h*)p;
        B1.h[1] = *(const v8h*)(p + 16);
        c1 = __builtin_amdgcn_wmma_f32_16x16x32_f16(
                 false, A1[ks].v, false, B1.v, (short)0, c1, false, false);
      }
      float bv = b1[being * MAX_H + hh + chunk * 16 + m];
      #pragma unroll
      for (int r = 0; r < 8; ++r) {
        float g = gelu_tanh(c1[r] + bv);
        hb[(r + hi * 8) * 40 + chunk * 16 + m] = (_Float16)g;  // C->row-major
      }
    }

    // ---- h tile as A fragment (wave-local LDS transpose) ----
    Frag16 A2;
    {
      const _Float16* p = hb + m * 40 + hi * 8;
      A2.h[0] = *(const v8h*)p;
      A2.h[1] = *(const v8h*)(p + 16);
    }

    // ---- GEMM2: accumulate y over this 32-wide H window ----
    #pragma unroll
    for (int nt = 0; nt < KT; ++nt) {
      Frag16 B2;
      const _Float16* p = w2t + (nt * 16 + m) * 40 + hi * 8;
      B2.h[0] = *(const v8h*)p;
      B2.h[1] = *(const v8h*)(p + 16);
      acc[nt] = __builtin_amdgcn_wmma_f32_16x16x32_f16(
                    false, A2.v, false, B2.v, (short)0, acc[nt], false, false);
    }
    __syncthreads();   // before restaging w1t/w2t
  }

  // ---- epilogue: +b2, scatter-add into scat via atomics ----
  #pragma unroll
  for (int nt = 0; nt < KT; ++nt) {
    int cl = nt * 16 + m;
    if (cl < K) {
      int col = sidx[cl];
      float bias = b2[being * KMAX + cl];
      #pragma unroll
      for (int r = 0; r < 8; ++r) {
        int row = row0 + wave * 16 + r + hi * 8;
        atomicAdd(&scat[row * NUM_BITS + col], acc[nt][r] + bias);
      }
    }
  }
}

__global__ void cov_kernel(const int* __restrict__ idx,
                           const int* __restrict__ step_p,
                           float* __restrict__ cov) {
  int being = blockIdx.x;
  if (step_p[0] % c_PER[being] != 0) return;
  int c = threadIdx.x;
  if (c < c_KS[being]) atomicAdd(&cov[idx[being * KMAX + c]], 1.0f);
}

__global__ void finalize_kernel(const float* __restrict__ x,
                                const float* __restrict__ scat,
                                const float* __restrict__ cov,
                                float* __restrict__ out) {
  int j = blockIdx.x * 256 + threadIdx.x;
  if (j < BATCH * NUM_BITS) {
    int col = j & (NUM_BITS - 1);
    out[j] = x[j] + scat[j] / fmaxf(cov[col], 1.0f);
  }
}

static inline size_t smem_bytes(int kpad) {
  int xs = kpad + 8;
  return 512 + (size_t)(128 + 32) * xs * 2     // xg + w1t
       + (size_t)kpad * 40 * 2                 // w2t
       + (size_t)8 * 16 * 40 * 2;              // per-wave h buffers
}

extern "C" void kernel_launch(void* const* d_in, const int* in_sizes, int n_in,
                              void* d_out, int out_size, void* d_ws, size_t ws_size,
                              hipStream_t stream) {
  const float* x   = (const float*)d_in[0];
  const float* W1  = (const float*)d_in[1];
  const float* b1  = (const float*)d_in[2];
  const float* W2  = (const float*)d_in[3];
  const float* b2  = (const float*)d_in[4];
  const int*   idx = (const int*)d_in[5];
  const int*   stp = (const int*)d_in[6];

  float* scat = (float*)d_ws;                    // 4096*256 f32
  float* cov  = scat + BATCH * NUM_BITS;         // 256 f32
  float* out  = (float*)d_out;

  hipMemsetAsync(d_ws, 0, (size_t)(BATCH * NUM_BITS + NUM_BITS) * sizeof(float),
                 stream);

  cov_kernel<<<NUM_BEINGS, KMAX, 0, stream>>>(idx, stp, cov);

  swarm_gemm<128><<<dim3(BATCH / 128, 4), 256, smem_bytes(128), stream>>>(
      x, W1, b1, W2, b2, idx, stp, scat);
  swarm_gemm<64><<<dim3(BATCH / 128, 4), 256, smem_bytes(64), stream>>>(
      x, W1, b1, W2, b2, idx, stp, scat);
  swarm_gemm<32><<<dim3(BATCH / 128, 24), 256, smem_bytes(32), stream>>>(
      x, W1, b1, W2, b2, idx, stp, scat);

  finalize_kernel<<<(BATCH * NUM_BITS) / 256, 256, 0, stream>>>(x, scat, cov, out);
}
